// HAN_1425929143039
// MI455X (gfx1250) — matrix operand order repaired and checked by reading, not compiled, not run on hardware
//
#include <hip/hip_runtime.h>
#include <cmath>

// ---------------- problem constants (fixed by the reference) ----------------
namespace {
constexpr int kN    = 50000;   // nodes
constexpr int kE    = 800000;  // edges per metapath
constexpr int kP    = 4;       // metapaths
constexpr int kIn   = 256;     // in feature dim
constexpr int kD    = 256;     // out feature dim
constexpr int kSemH = 128;     // semantic attention hidden

// workspace layout (bytes, 256-aligned)
constexpr size_t OFF_HF16 = 0;                                 // N*kIn*2      = 25,600,000
constexpr size_t OFF_WPK  = OFF_HF16 + (size_t)kN * kIn * 2;   // P*8*16*256*4 = 524,288
constexpr size_t OFF_W1PK = OFF_WPK + (size_t)kP * 8 * 16 * 256 * 4; // 8*8*256*4 = 65,536
constexpr size_t OFF_ROUT = OFF_W1PK + (size_t)8 * 8 * 256 * 4;      // P*N*4
constexpr size_t OFF_RIN  = OFF_ROUT + (size_t)kP * kN * 4;
constexpr size_t OFF_SSUM = OFF_RIN + (size_t)kP * kN * 4;           // 256 B (ssum[4], beta[4])
constexpr size_t OFF_X    = OFF_SSUM + 256;                          // N*D*4 (reused per metapath)
constexpr size_t OFF_Z    = OFF_X + (size_t)kN * kD * 4;             // P*N*D*4
} // namespace

typedef __attribute__((ext_vector_type(16))) _Float16 v16h;
typedef __attribute__((ext_vector_type(8)))  float    v8f;

union HPack { int4 q[2]; v16h v; };

__device__ __forceinline__ unsigned pack2_f16(float a, float b) {
  union { _Float16 h[2]; unsigned u; } u_;
  u_.h[0] = (_Float16)a;
  u_.h[1] = (_Float16)b;
  return u_.u;
}

// Async copy 16B from global to LDS (ASYNCcnt-tracked, bypasses VGPRs).
// ldsOff is the byte offset within this workgroup's LDS allocation
// (dynamic shared starts at LDS offset 0; HW adds LDS_BASE).
__device__ __forceinline__ void async_g2l_b128(unsigned ldsOff, const void* gaddr) {
  unsigned long long ga = (unsigned long long)gaddr;
  asm volatile("global_load_async_to_lds_b128 %0, %1, off"
               :: "v"(ldsOff), "v"(ga) : "memory");
}
__device__ __forceinline__ void wait_async0() {
  asm volatile("s_wait_asynccnt 0x0" ::: "memory");
}

// ---------------- conversion / packing ----------------
__global__ void k_cvt_h(const float* __restrict__ h, _Float16* __restrict__ hf16) {
  int idx = blockIdx.x * 256 + threadIdx.x;
  if (idx < kN * kIn) hf16[idx] = (_Float16)h[idx];
}

// Pack W_gc [P][kIn][kD] (row-major f32) into WMMA-B f16 layout:
// dword index = ((p*8 + kc)*16 + nt)*256 + lane*8 + v
// B element (lane,v,h): col = nt*16 + (lane&15), k = kc*32 + 16*(lane>=16) + 2*v + h
__global__ void k_pack_w(const float* __restrict__ Wgc, unsigned* __restrict__ wpk) {
  int idx = blockIdx.x * 256 + threadIdx.x;
  if (idx >= kP * 8 * 16 * 256) return;
  int v    = idx & 7;
  int lane = (idx >> 3) & 31;
  int nt   = (idx >> 8) & 15;
  int kc   = (idx >> 12) & 7;
  int p    = idx >> 15;
  int col  = nt * 16 + (lane & 15);
  int k    = kc * 32 + (lane >> 4) * 16 + v * 2;
  const float* wp = Wgc + (size_t)p * kIn * kD;
  wpk[idx] = pack2_f16(wp[(size_t)k * kD + col], wp[(size_t)(k + 1) * kD + col]);
}

// Pack W1 [kIn=D][kSemH] into WMMA-B f16 layout: dword = (kc*8+nt)*256 + lane*8 + v
__global__ void k_pack_w1(const float* __restrict__ W1, unsigned* __restrict__ w1pk) {
  int idx = blockIdx.x * 256 + threadIdx.x;
  if (idx >= 8 * 8 * 256) return;
  int v    = idx & 7;
  int lane = (idx >> 3) & 31;
  int nt   = (idx >> 8) & 7;
  int kc   = idx >> 11;
  int col  = nt * 16 + (lane & 15);
  int k    = kc * 32 + (lane >> 4) * 16 + v * 2;
  w1pk[idx] = pack2_f16(W1[(size_t)k * kSemH + col], W1[(size_t)(k + 1) * kSemH + col]);
}

// ---------------- degrees ----------------
__global__ void k_deg(const int* __restrict__ src, const int* __restrict__ dst,
                      float* __restrict__ dout, float* __restrict__ din) {
  int idx = blockIdx.x * 256 + threadIdx.x;
  if (idx >= kP * kE) return;
  int p = idx / kE;
  unsafeAtomicAdd(&dout[p * kN + src[idx]], 1.0f);
  unsafeAtomicAdd(&din[p * kN + dst[idx]], 1.0f);
}

__global__ void k_rsq(float* __restrict__ rout, float* __restrict__ rin) {
  int idx = blockIdx.x * 256 + threadIdx.x;
  if (idx >= kP * kN) return;
  rout[idx] = rsqrtf(fmaxf(rout[idx], 1.0f));
  rin[idx]  = rsqrtf(fmaxf(rin[idx], 1.0f));
}

// ---------------- GraphConv GEMM: x = (h @ W_p) * rsqrt(deg_out) ----------------
// Block = 8 waves sharing one (slab, 8 M-tiles). The 64KB packed-B slab
// (8 kc x 8 nt tiles) is async-DMA'd into LDS once and shared by all waves.
__global__ __launch_bounds__(256) void k_gemm(const _Float16* __restrict__ hf16,
                                              const unsigned* __restrict__ wpk,
                                              const float* __restrict__ rout,
                                              float* __restrict__ x) {
  extern __shared__ unsigned lds[];          // 16384 dwords = 64KB
  const int tid    = threadIdx.x;
  const int slab   = blockIdx.x & 1;
  const int mgroup = blockIdx.x >> 1;

  // stage B slab: dst dword (kc*8+nt)*256 + j  <-  src dword (kc*16 + slab*8 + nt)*256 + j
#pragma unroll
  for (int pass = 0; pass < 16; ++pass) {
    const int di   = pass * 1024 + tid * 4;  // dst dword index (b128 granularity)
    const int tile = di >> 8;                // 0..63
    const int j    = di & 255;
    const int kc   = tile >> 3;
    const int nt   = tile & 7;
    const int si   = (kc * 16 + slab * 8 + nt) * 256 + j;
    async_g2l_b128((unsigned)(di * 4), wpk + si);
  }
  wait_async0();
  __syncthreads();

  const int wave  = tid >> 5;
  const int mtile = mgroup * 8 + wave;
  if (mtile < kN / 16) {
    const int lane = tid & 31;
    const int hs   = lane >> 4;
    const int lrow = lane & 15;
    const _Float16* arow = hf16 + (size_t)(mtile * 16 + lrow) * kIn;

    v8f acc[8] = {};
    for (int kc = 0; kc < 8; ++kc) {
      HPack A;
      A.q[0] = *(const int4*)(arow + kc * 32 + hs * 16);
      A.q[1] = *(const int4*)(arow + kc * 32 + hs * 16 + 8);
#pragma unroll
      for (int nt = 0; nt < 8; ++nt) {
        const unsigned* bp = lds + (kc * 8 + nt) * 256 + lane * 8;
        HPack B;
        B.q[0] = *(const int4*)(bp);
        B.q[1] = *(const int4*)(bp + 4);
        acc[nt] = __builtin_amdgcn_wmma_f32_16x16x32_f16(false, A.v, false, B.v,
                                                         (short)0, acc[nt], false, false);
      }
    }
    float rs[8];
#pragma unroll
    for (int v = 0; v < 8; ++v) rs[v] = rout[mtile * 16 + hs * 8 + v];
#pragma unroll
    for (int nt = 0; nt < 8; ++nt) {
      const int col = (slab * 8 + nt) * 16 + lrow;
#pragma unroll
      for (int v = 0; v < 8; ++v)
        x[(size_t)(mtile * 16 + hs * 8 + v) * kD + col] = acc[nt][v] * rs[v];
    }
  }
}

// ---------------- edge scatter: z_p[dst] += x[src] ----------------
__global__ __launch_bounds__(256) void k_edge(const int* __restrict__ src,
                                              const int* __restrict__ dst,
                                              const float* __restrict__ x,
                                              float* __restrict__ zp) {
  const int t = threadIdx.x;
  const int base = blockIdx.x * 8;
#pragma unroll
  for (int i = 0; i < 8; ++i) {
    const int e = base + i;
    const int s = src[e];
    const int d = dst[e];
    unsafeAtomicAdd(&zp[(size_t)d * kD + t], x[(size_t)s * kD + t]);
  }
}

// ---------------- finalize: z = z * rsqrt(deg_in) + b ----------------
__global__ void k_fin(float* __restrict__ z, const float* __restrict__ rin,
                      const float* __restrict__ bgc) {
  int g = blockIdx.x * 256 + threadIdx.x;            // over P*N*(D/4)
  if (g >= kP * kN * (kD / 4)) return;
  int p  = g / (kN * 64);
  int r  = g % (kN * 64);
  int n  = r >> 6;
  int j4 = r & 63;
  float rv = rin[p * kN + n];
  float4 b = ((const float4*)bgc)[p * 64 + j4];
  float4* z4 = (float4*)z;
  float4 zz = z4[g];
  zz.x = zz.x * rv + b.x;
  zz.y = zz.y * rv + b.y;
  zz.z = zz.z * rv + b.z;
  zz.w = zz.w * rv + b.w;
  z4[g] = zz;
}

// ---------------- semantic attention scores: ssum[p] += sum_n tanh(z W1 + b1) . w2 ----
// Full packed W1 (64KB) async-staged into LDS, shared by all 8 waves.
__global__ __launch_bounds__(256) void k_attn(const float* __restrict__ z,
                                              const unsigned* __restrict__ w1pk,
                                              const float* __restrict__ b1,
                                              const float* __restrict__ w2,
                                              float* __restrict__ ssum) {
  extern __shared__ unsigned lds[];          // 16384 dwords = 64KB
  const int tid = threadIdx.x;
#pragma unroll
  for (int pass = 0; pass < 16; ++pass) {
    const int di = pass * 1024 + tid * 4;
    async_g2l_b128((unsigned)(di * 4), w1pk + di);
  }
  wait_async0();
  __syncthreads();

  const int wave = blockIdx.x * 8 + (tid >> 5);
  if (wave >= kP * (kN / 16)) return;
  const int lane  = tid & 31;
  const int p     = wave / (kN / 16);
  const int mtile = wave % (kN / 16);
  const int hs    = lane >> 4;
  const int lrow  = lane & 15;
  const float* zrow = z + ((size_t)p * kN + mtile * 16 + lrow) * kD;

  v8f acc[8] = {};
  for (int kc = 0; kc < 8; ++kc) {
    const float* zp = zrow + kc * 32 + hs * 16;
    float4 f0 = *(const float4*)(zp);
    float4 f1 = *(const float4*)(zp + 4);
    float4 f2 = *(const float4*)(zp + 8);
    float4 f3 = *(const float4*)(zp + 12);
    v16h a;
    a[0]  = (_Float16)f0.x; a[1]  = (_Float16)f0.y; a[2]  = (_Float16)f0.z; a[3]  = (_Float16)f0.w;
    a[4]  = (_Float16)f1.x; a[5]  = (_Float16)f1.y; a[6]  = (_Float16)f1.z; a[7]  = (_Float16)f1.w;
    a[8]  = (_Float16)f2.x; a[9]  = (_Float16)f2.y; a[10] = (_Float16)f2.z; a[11] = (_Float16)f2.w;
    a[12] = (_Float16)f3.x; a[13] = (_Float16)f3.y; a[14] = (_Float16)f3.z; a[15] = (_Float16)f3.w;
#pragma unroll
    for (int nt = 0; nt < 8; ++nt) {
      const unsigned* bp = lds + (kc * 8 + nt) * 256 + lane * 8;
      HPack B;
      B.q[0] = *(const int4*)(bp);
      B.q[1] = *(const int4*)(bp + 4);
      acc[nt] = __builtin_amdgcn_wmma_f32_16x16x32_f16(false, a, false, B.v,
                                                       (short)0, acc[nt], false, false);
    }
  }
  float partial = 0.0f;
#pragma unroll
  for (int nt = 0; nt < 8; ++nt) {
    const int col = nt * 16 + lrow;
    const float bb = b1[col];
    const float ww = w2[col];
#pragma unroll
    for (int v = 0; v < 8; ++v)
      partial += tanhf(acc[nt][v] + bb) * ww;
  }
#pragma unroll
  for (int off = 16; off >= 1; off >>= 1)
    partial += __shfl_xor(partial, off, 32);
  if (lane == 0) unsafeAtomicAdd(&ssum[p], partial);
}

// ---------------- softmax over metapaths ----------------
__global__ void k_beta(const float* __restrict__ ssum, float* __restrict__ beta) {
  if (threadIdx.x != 0 || blockIdx.x != 0) return;
  float s[kP];
  float m = -1e30f;
  for (int p = 0; p < kP; ++p) { s[p] = ssum[p] / (float)kN; m = fmaxf(m, s[p]); }
  float tot = 0.0f;
  for (int p = 0; p < kP; ++p) { s[p] = expf(s[p] - m); tot += s[p]; }
  for (int p = 0; p < kP; ++p) beta[p] = s[p] / tot;
}

// ---------------- output: out = sum_p beta[p] * z_p ----------------
__global__ void k_out(const float* __restrict__ z, const float* __restrict__ beta,
                      float* __restrict__ out) {
  int g = blockIdx.x * 256 + threadIdx.x;            // over N*(D/4)
  if (g >= kN * 64) return;
  const float b0 = beta[0], b1_ = beta[1], b2 = beta[2], b3 = beta[3];
  const float4* z4 = (const float4*)z;
  const size_t st = (size_t)kN * 64;
  float4 a0 = z4[g], a1 = z4[g + st], a2 = z4[g + 2 * st], a3 = z4[g + 3 * st];
  float4 o;
  o.x = b0 * a0.x + b1_ * a1.x + b2 * a2.x + b3 * a3.x;
  o.y = b0 * a0.y + b1_ * a1.y + b2 * a2.y + b3 * a3.y;
  o.z = b0 * a0.z + b1_ * a1.z + b2 * a2.z + b3 * a3.z;
  o.w = b0 * a0.w + b1_ * a1.w + b2 * a2.w + b3 * a3.w;
  ((float4*)out)[g] = o;
}

extern "C" void kernel_launch(void* const* d_in, const int* in_sizes, int n_in,
                              void* d_out, int out_size, void* d_ws, size_t ws_size,
                              hipStream_t stream) {
  (void)in_sizes; (void)n_in; (void)out_size; (void)ws_size;
  const float* h   = (const float*)d_in[0];
  const int*   src = (const int*)d_in[1];
  const int*   dst = (const int*)d_in[2];
  const float* Wgc = (const float*)d_in[3];
  const float* bgc = (const float*)d_in[4];
  const float* W1  = (const float*)d_in[5];
  const float* b1  = (const float*)d_in[6];
  const float* w2  = (const float*)d_in[7];
  float* out = (float*)d_out;
  char* ws = (char*)d_ws;

  _Float16* hf16 = (_Float16*)(ws + OFF_HF16);
  unsigned* wpk  = (unsigned*)(ws + OFF_WPK);
  unsigned* w1pk = (unsigned*)(ws + OFF_W1PK);
  float* rout = (float*)(ws + OFF_ROUT);
  float* rin  = (float*)(ws + OFF_RIN);
  float* ssum = (float*)(ws + OFF_SSUM);
  float* beta = ssum + 4;
  float* x    = (float*)(ws + OFF_X);
  float* z    = (float*)(ws + OFF_Z);

  hipMemsetAsync(rout, 0, (size_t)kP * kN * 4, stream);
  hipMemsetAsync(rin,  0, (size_t)kP * kN * 4, stream);
  hipMemsetAsync(ssum, 0, 32, stream);
  hipMemsetAsync(z,    0, (size_t)kP * kN * kD * 4, stream);

  k_cvt_h  <<<(kN * kIn) / 256, 256, 0, stream>>>(h, hf16);
  k_pack_w <<<(kP * 8 * 16 * 256) / 256, 256, 0, stream>>>(Wgc, wpk);
  k_pack_w1<<<(8 * 8 * 256) / 256, 256, 0, stream>>>(W1, w1pk);
  k_deg    <<<(kP * kE) / 256, 256, 0, stream>>>(src, dst, rout, rin);
  k_rsq    <<<(kP * kN + 255) / 256, 256, 0, stream>>>(rout, rin);

  const int mgroups = (kN / 16 + 7) / 8;             // 391
  for (int p = 0; p < kP; ++p) {
    k_gemm<<<mgroups * 2, 256, 65536, stream>>>(
        hf16, wpk + (size_t)p * (8 * 16 * 256), rout + (size_t)p * kN, x);
    k_edge<<<kE / 8, 256, 0, stream>>>(
        src + (size_t)p * kE, dst + (size_t)p * kE, x, z + (size_t)p * kN * kD);
  }

  k_fin <<<(kP * kN * 64) / 256, 256, 0, stream>>>(z, rin, bgc);
  const int attnWaves = kP * (kN / 16);              // 12500
  k_attn<<<(attnWaves + 7) / 8, 256, 65536, stream>>>(z, w1pk, b1, w2, ssum);
  k_beta<<<1, 32, 0, stream>>>(ssum, beta);
  k_out <<<(kN * 64) / 256, 256, 0, stream>>>(z, beta, out);
}
